// MultiheadAttention_74766790688795
// MI455X (gfx1250) — compile-verified, matrix operand order checked by reading, not run on hardware
//
#include <hip/hip_runtime.h>
#include <hip/hip_bf16.h>
#include <math.h>

typedef __attribute__((ext_vector_type(16))) __bf16       v16bf;
typedef __attribute__((ext_vector_type(8)))  float        v8f;
typedef __attribute__((ext_vector_type(4)))  unsigned int v4u;

union FragU { v16bf v; v4u q[2]; };

constexpr int cB = 2, cLQ = 2048, cLK = 2048, cC = 1024, cDK = 1024, cDV = 1024, cH = 16, cDH = 64;

__device__ inline v8f wmma_bf16(v16bf a, v16bf b, v8f c) {
  // D = A(16x32 bf16) * B(32x16 bf16) + C(16x16 f32)
  return __builtin_amdgcn_wmma_f32_16x16x32_bf16(false, a, false, b, (short)0, c, false, false);
}

// Load a 16(row) x 32(k) bf16 fragment in WMMA A/B register layout from a
// row-major matrix. p0 points at [row0, k0]; ld in elements; k0 multiple of 32.
__device__ inline v16bf load_frag(const __bf16* p0, int ld, int lane) {
  int r  = lane & 15;
  int kh = (lane & 16) ? 8 : 0;
  const __bf16* p = p0 + (size_t)r * ld + kh;
  FragU u;
  u.q[0] = *(const v4u*)(p);
  u.q[1] = *(const v4u*)(p + 16);
  return u.v;
}

__global__ void cvt_f32_bf16(const float* __restrict__ src, __bf16* __restrict__ dst, int n) {
  int i = blockIdx.x * blockDim.x + threadIdx.x;
  if (i < n) dst[i] = (__bf16)src[i];
}

// D = (A @ Bt^T + bias[n]) * scale
// A : M x K row-major bf16 (lda); Bt : N x K row-major bf16 (ldb)
// OMODE 0: f32 out [m*ldo+n]; 1: bf16 out [m*ldo+n]; 2: bf16 out transposed [n*ldo+m]
template <int OMODE>
__global__ __launch_bounds__(128) void gemm_nt(const __bf16* __restrict__ A,
                                               const __bf16* __restrict__ Bt,
                                               const float* __restrict__ bias,
                                               void* __restrict__ out,
                                               int K, int lda, int ldb, int ldo, float scale) {
  int lane = threadIdx.x & 31;
  int wv   = threadIdx.x >> 5;
  int m0 = blockIdx.y * 64 + (wv >> 1) * 32;
  int n0 = blockIdx.x * 64 + (wv & 1) * 32;
  v8f acc[2][2] = {};
  for (int kk = 0; kk < K; kk += 32) {
    v16bf a0 = load_frag(A + (size_t)m0 * lda + kk, lda, lane);
    v16bf a1 = load_frag(A + (size_t)(m0 + 16) * lda + kk, lda, lane);
    v16bf b0 = load_frag(Bt + (size_t)n0 * ldb + kk, ldb, lane);
    v16bf b1 = load_frag(Bt + (size_t)(n0 + 16) * ldb + kk, ldb, lane);
    acc[0][0] = wmma_bf16(a0, b0, acc[0][0]);
    acc[0][1] = wmma_bf16(a0, b1, acc[0][1]);
    acc[1][0] = wmma_bf16(a1, b0, acc[1][0]);
    acc[1][1] = wmma_bf16(a1, b1, acc[1][1]);
  }
  int half = lane >> 4, nl = lane & 15;
  #pragma unroll
  for (int mi = 0; mi < 2; mi++) {
    #pragma unroll
    for (int nj = 0; nj < 2; nj++) {
      int n = n0 + nj * 16 + nl;
      float bn = bias[n];
      #pragma unroll
      for (int v = 0; v < 8; v++) {
        int m = m0 + mi * 16 + v + 8 * half;
        float val = (acc[mi][nj][v] + bn) * scale;
        if (OMODE == 0)      ((float*)out)[(size_t)m * ldo + n] = val;
        else if (OMODE == 1) ((__bf16*)out)[(size_t)m * ldo + n] = (__bf16)val;
        else                 ((__bf16*)out)[(size_t)n * ldo + m] = (__bf16)val;
      }
    }
  }
}

// One wave per (b, h, 16-row q tile). Flash-attention streaming over LK in
// 128-key blocks so the cross-lane softmax reductions amortize over 16 WMMAs.
__global__ __launch_bounds__(128) void attn_kernel(const __bf16* __restrict__ qbf,
                                                   const __bf16* __restrict__ kbf,
                                                   const __bf16* __restrict__ vT,
                                                   const float* __restrict__ wmat,
                                                   const float* __restrict__ bias,
                                                   __bf16* __restrict__ xbf,
                                                   float* __restrict__ top) {
  __shared__ __align__(16) __bf16 plds[4][16 * 128];
  int lane = threadIdx.x & 31;
  int wv   = threadIdx.x >> 5;
  int gw = blockIdx.x * 4 + wv;
  int qt = gw & (cLQ / 16 - 1);          // 7 bits
  int h  = (gw >> 7) & (cH - 1);         // 4 bits
  int b  = gw >> 11;
  int q0 = qt * 16;
  int half = lane >> 4, nl = lane & 15;
  int kh = half ? 8 : 0;

  const __bf16* qrow = qbf + ((size_t)(b * cLQ + q0)) * cDK + h * cDH;
  v16bf qa0 = load_frag(qrow, cDK, lane);        // d = 0..31
  v16bf qa1 = load_frag(qrow + 32, cDK, lane);   // d = 32..63

  const float*  wrow  = wmat + ((size_t)b * cLQ + q0) * cLK;
  const float*  brow  = bias + (size_t)b * cLK;
  const __bf16* kbase = kbf + ((size_t)b * cLK) * cDK + h * cDH;

  float m8[8], l8[8];
  v8f o[4] = {};
  #pragma unroll
  for (int v = 0; v < 8; v++) { m8[v] = -1e30f; l8[v] = 0.0f; }

  __bf16* L = plds[wv];

  for (int k0 = 0; k0 < cLK; k0 += 128) {
    // ---- logits: 8 tiles of 16x16 (keys k0 .. k0+127) ----
    v8f s[8];
    #pragma unroll
    for (int t = 0; t < 8; t++) {
      const __bf16* kp = kbase + (size_t)(k0 + t * 16) * cDK;
      v8f z = {};
      z = wmma_bf16(qa0, load_frag(kp, cDK, lane), z);
      z = wmma_bf16(qa1, load_frag(kp + 32, cDK, lane), z);
      s[t] = z;
    }
    float bb[8];
    #pragma unroll
    for (int t = 0; t < 8; t++) bb[t] = brow[k0 + t * 16 + nl];

    // ---- weight * logits + bias, online softmax with amortized reductions ----
    #pragma unroll
    for (int v = 0; v < 8; v++) {
      int M = v + 8 * half;
      const float* wr = wrow + (size_t)M * cLK + k0 + nl;
      float rm = -1e30f;
      #pragma unroll
      for (int t = 0; t < 8; t++) {
        float tv = s[t][v] * wr[t * 16] + bb[t];
        s[t][v] = tv;
        rm = fmaxf(rm, tv);
      }
      rm = fmaxf(rm, __shfl_xor(rm, 1, 32));
      rm = fmaxf(rm, __shfl_xor(rm, 2, 32));
      rm = fmaxf(rm, __shfl_xor(rm, 4, 32));
      rm = fmaxf(rm, __shfl_xor(rm, 8, 32));
      float mnew = fmaxf(m8[v], rm);
      float c = __expf(m8[v] - mnew);
      float rs = 0.0f;
      #pragma unroll
      for (int t = 0; t < 8; t++) {
        float e = __expf(s[t][v] - mnew);
        s[t][v] = e;
        rs += e;
      }
      rs += __shfl_xor(rs, 1, 32);
      rs += __shfl_xor(rs, 2, 32);
      rs += __shfl_xor(rs, 4, 32);
      rs += __shfl_xor(rs, 8, 32);
      l8[v] = l8[v] * c + rs;
      m8[v] = mnew;
      #pragma unroll
      for (int j = 0; j < 4; j++) o[j][v] = o[j][v] * c;
      // stage P (row M, 128 keys) into LDS for the A-frag reloads
      #pragma unroll
      for (int t = 0; t < 8; t++) L[M * 128 + t * 16 + nl] = (__bf16)s[t][v];
    }

    // ---- P @ V : 4 gemm-k chunks of 32 keys, 4 d-frags each ----
    #pragma unroll
    for (int kc = 0; kc < 4; kc++) {
      FragU pu;
      pu.q[0] = *(const v4u*)&L[nl * 128 + kc * 32 + kh];
      pu.q[1] = *(const v4u*)&L[nl * 128 + kc * 32 + kh + 16];
      v16bf pa = pu.v;
      #pragma unroll
      for (int j = 0; j < 4; j++) {
        v16bf vb = load_frag(vT + ((size_t)(h * cDH + j * 16)) * (cB * cLK) + b * cLK + k0 + kc * 32,
                             cB * cLK, lane);
        o[j] = wmma_bf16(pa, vb, o[j]);
      }
    }
  }

  float inv[8];
  #pragma unroll
  for (int v = 0; v < 8; v++) inv[v] = 1.0f / l8[v];

  // x laid out as (b, lq, h*64+d) == reference transpose+reshape
  __bf16* xr = xbf + ((size_t)(b * cLQ + q0)) * cDV + h * cDH;
  #pragma unroll
  for (int j = 0; j < 4; j++)
    #pragma unroll
    for (int v = 0; v < 8; v++)
      xr[(size_t)(v + 8 * half) * cDV + j * 16 + nl] = (__bf16)(o[j][v] * inv[v]);

  if (h == 0) {  // second pass: emit normalized probabilities for top_attn
    float* tr = top + ((size_t)b * cLQ + q0) * cLK;
    for (int k0 = 0; k0 < cLK; k0 += 32) {
      v8f s0 = {}, s1 = {};
      s0 = wmma_bf16(qa0, load_frag(kbase + (size_t)k0 * cDK, cDK, lane), s0);
      s0 = wmma_bf16(qa1, load_frag(kbase + (size_t)k0 * cDK + 32, cDK, lane), s0);
      s1 = wmma_bf16(qa0, load_frag(kbase + (size_t)(k0 + 16) * cDK, cDK, lane), s1);
      s1 = wmma_bf16(qa1, load_frag(kbase + (size_t)(k0 + 16) * cDK + 32, cDK, lane), s1);
      float bb0 = brow[k0 + nl], bb1 = brow[k0 + 16 + nl];
      #pragma unroll
      for (int v = 0; v < 8; v++) {
        int M = v + 8 * half;
        float t0 = s0[v] * wrow[(size_t)M * cLK + k0 + nl] + bb0;
        float t1 = s1[v] * wrow[(size_t)M * cLK + k0 + 16 + nl] + bb1;
        tr[(size_t)M * cLK + k0 + nl]      = __expf(t0 - m8[v]) * inv[v];
        tr[(size_t)M * cLK + k0 + 16 + nl] = __expf(t1 - m8[v]) * inv[v];
      }
    }
  }
}

extern "C" void kernel_launch(void* const* d_in, const int* in_sizes, int n_in,
                              void* d_out, int out_size, void* d_ws, size_t ws_size,
                              hipStream_t stream) {
  (void)in_sizes; (void)n_in; (void)out_size; (void)ws_size;
  const float* query  = (const float*)d_in[0];
  const float* memory = (const float*)d_in[1];
  // d_in[2] = num_heads (hardcoded 16)
  const float* bias   = (const float*)d_in[3];
  const float* wmat   = (const float*)d_in[4];
  const float* Wq = (const float*)d_in[5];
  const float* bq = (const float*)d_in[6];
  const float* Wk = (const float*)d_in[7];
  const float* bk = (const float*)d_in[8];
  const float* Wv = (const float*)d_in[9];
  const float* bv = (const float*)d_in[10];
  const float* Wo = (const float*)d_in[11];
  const float* bo = (const float*)d_in[12];

  char* ws = (char*)d_ws;
  size_t off = 0;
  auto alloc = [&](size_t bytes) {
    char* p = ws + off;
    off += (bytes + 255) & ~(size_t)255;
    return p;
  };
  __bf16* qrybf = (__bf16*)alloc((size_t)cB * cLQ * cC * 2);
  __bf16* membf = (__bf16*)alloc((size_t)cB * cLK * cC * 2);
  __bf16* wqbf  = (__bf16*)alloc((size_t)cDK * cC * 2);
  __bf16* wkbf  = (__bf16*)alloc((size_t)cDK * cC * 2);
  __bf16* wvbf  = (__bf16*)alloc((size_t)cDV * cC * 2);
  __bf16* wobf  = (__bf16*)alloc((size_t)cC * cDV * 2);
  __bf16* qbf   = (__bf16*)alloc((size_t)cB * cLQ * cDK * 2);
  __bf16* kbf   = (__bf16*)alloc((size_t)cB * cLK * cDK * 2);
  __bf16* vTbf  = (__bf16*)alloc((size_t)cDV * cB * cLK * 2);   // [d][b*LK+k]
  __bf16* xbf   = (__bf16*)alloc((size_t)cB * cLQ * cDV * 2);

  auto cvt = [&](const float* s, __bf16* d, int n) {
    cvt_f32_bf16<<<(n + 255) / 256, 256, 0, stream>>>(s, d, n);
  };
  cvt(query,  qrybf, cB * cLQ * cC);
  cvt(memory, membf, cB * cLK * cC);
  cvt(Wq, wqbf, cDK * cC);
  cvt(Wk, wkbf, cDK * cC);
  cvt(Wv, wvbf, cDV * cC);
  cvt(Wo, wobf, cC * cDV);

  dim3 blk(128);
  dim3 gp(cDK / 64, (cB * cLQ) / 64);
  // q = (query @ Wq^T + bq) / sqrt(64)
  gemm_nt<1><<<gp, blk, 0, stream>>>(qrybf, wqbf, bq, qbf, cC, cC, cC, cDK, 0.125f);
  gemm_nt<1><<<gp, blk, 0, stream>>>(membf, wkbf, bk, kbf, cC, cC, cC, cDK, 1.0f);
  gemm_nt<2><<<gp, blk, 0, stream>>>(membf, wvbf, bv, vTbf, cC, cC, cC, cB * cLK, 1.0f);

  float* out = (float*)d_out;
  float* top = out + (size_t)cB * cLQ * cC;
  attn_kernel<<<(cB * cH * (cLQ / 16)) / 4, blk, 0, stream>>>(qbf, kbf, vTbf, wmat, bias, xbf, top);

  dim3 go(cC / 64, (cB * cLQ) / 64);
  gemm_nt<0><<<go, blk, 0, stream>>>(xbf, wobf, bo, out, cDV, cDV, cDV, cC, 1.0f);
}
